// SGNHeadOcc_5789615915174
// MI455X (gfx1250) — matrix operand order repaired and checked by reading, not compiled, not run on hardware
//
#include <hip/hip_runtime.h>
#include <hip/hip_bf16.h>

// ---------------------------------------------------------------------------
// Types for CDNA5 WMMA (wave32): D(16x16 f32) = A(16x32 bf16) x B(32x16 bf16) + C
// ---------------------------------------------------------------------------
typedef __bf16 bf16_t;
typedef __attribute__((ext_vector_type(16))) __bf16 v16bf;
typedef __attribute__((ext_vector_type(8)))  float  v8f;

#define N_PTS 500000

// Padded (to 16) segment-table geometry for P_SCALES = (2,4,6,8), SPATIAL=(256,256,32)
//  ps=2: 128*128*16 = 262144   ps=4: 64*64*8 = 32768
//  ps=6: 43*43*6    = 11094 -> pad 11104    ps=8: 32*32*4 = 4096
#define SEG0 0
#define SEG1 262144
#define SEG2 294912
#define SEG3 306016
#define TOTSEG 310112                 // divisible by 16

// workspace layout (float units)
#define RED_OFF 0LL                               // N*64        = 32,000,000
#define SUM_OFF 32000000LL                        // TOTSEG*64   = 19,847,168
#define CNT_OFF (SUM_OFF + (long long)TOTSEG*64)  // TOTSEG
#define ATT_OFF (CNT_OFF + (long long)TOTSEG)     // TOTSEG*32

__device__ __forceinline__ int segoff(int j) {
  const int o[4] = {SEG0, SEG1, SEG2, SEG3};
  return o[j];
}

__device__ __forceinline__ int seg_of(int x, int y, int z, int j) {
  const int ps[4] = {2, 4, 6, 8};
  const int gy[4] = {128, 64, 43, 32};
  const int gz[4] = {16, 8, 6, 4};
  int a = x / ps[j], b = y / ps[j], c = z / ps[j];
  return (a * gy[j] + b) * gz[j] + c;   // batch index is always 0
}

// --- fragment index helpers (per CDNA5 ISA 7.12.2, 16-bit A 16x32 / B 32x16) ---
__device__ __forceinline__ int a_col(int lane, int e) {
  int p = e >> 1;
  int base = (p < 4) ? (p << 1) : (16 + ((p - 4) << 1));
  return base + ((lane >> 4) << 3) + (e & 1);
}

// A fragment: one matrix row per lane (lanes L and L+16 share row L, differ in K)
__device__ __forceinline__ v16bf load_a_row(const float* __restrict__ rowp, int lane) {
  v16bf a;
#pragma unroll
  for (int e = 0; e < 16; ++e) a[e] = (bf16_t)rowp[a_col(lane, e)];
  return a;
}

__device__ __forceinline__ v16bf load_a_row_scaled(const float* __restrict__ rowp,
                                                   float s, int lane) {
  v16bf a;
#pragma unroll
  for (int e = 0; e < 16; ++e) a[e] = (bf16_t)(rowp[a_col(lane, e)] * s);
  return a;
}

// B fragment from row-major weight W[K][ld] : lane = column, K in table order (matches a_col)
__device__ __forceinline__ v16bf load_b_frag(const float* __restrict__ W, int ld,
                                             int kbase, int nbase, int lane) {
  int n = nbase + (lane & 15);
  int k0 = kbase + ((lane >> 4) << 3);
  v16bf b;
#pragma unroll
  for (int e = 0; e < 16; ++e) b[e] = (bf16_t)W[(k0 + e + ((e >> 3) << 3)) * ld + n];
  return b;
}

__device__ __forceinline__ v8f wmma_bf16(v16bf a, v16bf b, v8f c) {
  return __builtin_amdgcn_wmma_f32_16x16x32_bf16(false, a, false, b, (short)0, c,
                                                 false, false);
}

// ---------------------------------------------------------------------------
// Kernel 0: zero segment sums + counts
// ---------------------------------------------------------------------------
__global__ void k_zero(float* __restrict__ p, long long n) {
  long long i = (long long)blockIdx.x * blockDim.x + threadIdx.x;
  long long stride = (long long)gridDim.x * blockDim.x;
  for (; i < n; i += stride) p[i] = 0.0f;
}

// ---------------------------------------------------------------------------
// Kernel 1: red = relu(feats @ W_red + b_red), fused 4-scale atomic scatter
// one wave = one 16-row tile; 8 WMMA/wave
// ---------------------------------------------------------------------------
__global__ __launch_bounds__(256) void k_red_scatter(
    const float* __restrict__ feats, const int* __restrict__ coords,
    const float* __restrict__ W_red, const float* __restrict__ b_red,
    float* __restrict__ red, float* __restrict__ sums, float* __restrict__ cnt) {
  const int lane = threadIdx.x & 31;
  const int wave = threadIdx.x >> 5;
  const int tile = blockIdx.x * (blockDim.x >> 5) + wave;
  const int rowbase = tile * 16;
  if (rowbase >= N_PTS) return;  // no barriers in this kernel

  const int m16 = lane & 15;
  const int half = lane >> 4;
  int rA = rowbase + m16;
  if (rA >= N_PTS) rA = N_PTS - 1;

  v16bf a0 = load_a_row(feats + (long long)rA * 64, lane);
  v16bf a1 = load_a_row(feats + (long long)rA * 64 + 32, lane);

  int x = coords[rA * 4 + 1], y = coords[rA * 4 + 2], z = coords[rA * 4 + 3];
  int segs[4];
#pragma unroll
  for (int j = 0; j < 4; ++j) segs[j] = seg_of(x, y, z, j);

  // Hoisted cross-lane broadcast: lane handles rows t = r + 8*half in the C
  // fragment; pre-shuffle the (segoff-folded) segment ids once (nt-invariant).
  int sfull[4][8];
#pragma unroll
  for (int j = 0; j < 4; ++j)
#pragma unroll
    for (int r = 0; r < 8; ++r)
      sfull[j][r] = segoff(j) + __shfl(segs[j], r + 8 * half, 32);

  const bool fulltile = (rowbase + 16 <= N_PTS);

#pragma unroll
  for (int nt = 0; nt < 4; ++nt) {
    v16bf b0 = load_b_frag(W_red, 64, 0, nt * 16, lane);
    v16bf b1 = load_b_frag(W_red, 64, 32, nt * 16, lane);
    v8f c = {};
    c = wmma_bf16(a0, b0, c);
    c = wmma_bf16(a1, b1, c);
    const int n = nt * 16 + m16;
    const float bias = b_red[n];
    if (fulltile) {
#pragma unroll
      for (int r = 0; r < 8; ++r) {
        int m = rowbase + r + 8 * half;
        float v = c[r] + bias;
        v = v > 0.f ? v : 0.f;
        red[(long long)m * 64 + n] = v;
#pragma unroll
        for (int j = 0; j < 4; ++j)
          atomicAdd(&sums[(long long)sfull[j][r] * 64 + n], v);
      }
    } else {
#pragma unroll
      for (int r = 0; r < 8; ++r) {
        int m = rowbase + r + 8 * half;
        if (m >= N_PTS) continue;
        float v = c[r] + bias;
        v = v > 0.f ? v : 0.f;
        red[(long long)m * 64 + n] = v;
#pragma unroll
        for (int j = 0; j < 4; ++j)
          atomicAdd(&sums[(long long)sfull[j][r] * 64 + n], v);
      }
    }
  }
  if (lane < 16 && rowbase + lane < N_PTS) {
#pragma unroll
    for (int j = 0; j < 4; ++j) atomicAdd(&cnt[segoff(j) + segs[j]], 1.0f);
  }
}

// ---------------------------------------------------------------------------
// Kernel 2: per-segment mean + att = relu(fkm @ fc_list_W[j] + b[j])
// one wave = 16 segments; 4 WMMA/wave
// ---------------------------------------------------------------------------
__global__ __launch_bounds__(256) void k_seg_fc(
    const float* __restrict__ fc_list_W, const float* __restrict__ fc_list_b,
    const float* __restrict__ sums, const float* __restrict__ cnt,
    float* __restrict__ att) {
  const int lane = threadIdx.x & 31;
  const int wave = threadIdx.x >> 5;
  const int tile = blockIdx.x * (blockDim.x >> 5) + wave;
  const int segbase = tile * 16;
  if (segbase >= TOTSEG) return;

  int j = (segbase >= SEG3) ? 3 : (segbase >= SEG2) ? 2 : (segbase >= SEG1) ? 1 : 0;
  const int m16 = lane & 15;
  const int half = lane >> 4;
  const int s = segbase + m16;

  float inv = 1.0f / fmaxf(cnt[s], 1.0f);
  v16bf a0 = load_a_row_scaled(sums + (long long)s * 64, inv, lane);
  v16bf a1 = load_a_row_scaled(sums + (long long)s * 64 + 32, inv, lane);

  const float* Wj = fc_list_W + j * 64 * 32;
#pragma unroll
  for (int nt = 0; nt < 2; ++nt) {
    v16bf b0 = load_b_frag(Wj, 32, 0, nt * 16, lane);
    v16bf b1 = load_b_frag(Wj, 32, 32, nt * 16, lane);
    v8f c = {};
    c = wmma_bf16(a0, b0, c);
    c = wmma_bf16(a1, b1, c);
    const int n = nt * 16 + m16;
    const float bias = fc_list_b[j * 32 + n];
#pragma unroll
    for (int r = 0; r < 8; ++r) {
      int sm = segbase + r + 8 * half;
      float v = c[r] + bias;
      att[(long long)sm * 32 + n] = v > 0.f ? v : 0.f;
    }
  }
}

// ---------------------------------------------------------------------------
// Kernel 3: per-point attention fusion + output projection; 38 WMMA/wave
// 4 waves/block, per-wave LDS staging; no divergent exits (uniform barriers)
// ---------------------------------------------------------------------------
__global__ __launch_bounds__(128) void k_final(
    const int* __restrict__ coords, const float* __restrict__ fcs_W,
    const float* __restrict__ fcs_b, const float* __restrict__ fc_W,
    const float* __restrict__ out_fc_W, const float* __restrict__ lo1_W,
    const float* __restrict__ lo2_W, const float* __restrict__ lo2_b,
    const float* __restrict__ red, const float* __restrict__ att,
    float* __restrict__ out) {
  __shared__ __align__(16) float sA[4][16][128];  // scale_features, later out_f
  __shared__ __align__(16) float sB[4][16][64];   // feat_Z | sfw, later h

  const int lane = threadIdx.x & 31;
  const int w = threadIdx.x >> 5;
  const int tile = blockIdx.x * 4 + w;
  const int rowbase = tile * 16;
  const int m16 = lane & 15;
  const int half = lane >> 4;
  int prow = rowbase + m16;
  if (prow >= N_PTS) prow = N_PTS - 1;

  // 1. gather scale_features: sA[w][m][j*32+n]  (lanes split 2 scales each)
  int x = coords[prow * 4 + 1], y = coords[prow * 4 + 2], z = coords[prow * 4 + 3];
#pragma unroll
  for (int jj = 0; jj < 2; ++jj) {
    int j = 2 * half + jj;
    int sj = segoff(j) + seg_of(x, y, z, j);
    const float4* src = (const float4*)(att + (long long)sj * 32);
    float4* dst = (float4*)(&sA[w][m16][j * 32]);
#pragma unroll
    for (int q = 0; q < 8; ++q) dst[q] = src[q];
  }
  __syncthreads();

  // 2. feat_S = sum_j scale_features
  v16bf a_fs;
#pragma unroll
  for (int e = 0; e < 16; ++e) {
    int k = a_col(lane, e);
    a_fs[e] = (bf16_t)(sA[w][m16][k] + sA[w][m16][32 + k] + sA[w][m16][64 + k] +
                       sA[w][m16][96 + k]);
  }
  // 3. feat_Z = relu(feat_S @ fc_W) -> sB[:,0:32]
#pragma unroll
  for (int nt = 0; nt < 2; ++nt) {
    v16bf b = load_b_frag(fc_W, 32, 0, nt * 16, lane);
    v8f c = {};
    c = wmma_bf16(a_fs, b, c);
    int n = nt * 16 + m16;
#pragma unroll
    for (int r = 0; r < 8; ++r) {
      float v = c[r];
      sB[w][r + 8 * half][n] = v > 0.f ? v : 0.f;
    }
  }
  __syncthreads();

  // 4. A-frag of feat_Z
  v16bf a_fz;
#pragma unroll
  for (int e = 0; e < 16; ++e) a_fz[e] = (bf16_t)sB[w][m16][a_col(lane, e)];

  // 5. attn_j = sigmoid(feat_Z @ fcs_W[j] + b); sfw = sum_j s_j*attn_j -> sB[:,32:64]
#pragma unroll
  for (int nt = 0; nt < 2; ++nt) {
    int n = nt * 16 + m16;
    float sfw[8];
#pragma unroll
    for (int r = 0; r < 8; ++r) sfw[r] = 0.f;
#pragma unroll
    for (int j = 0; j < 4; ++j) {
      v16bf b = load_b_frag(fcs_W + j * 32 * 32, 32, 0, nt * 16, lane);
      v8f c = {};
      c = wmma_bf16(a_fz, b, c);
      float bias = fcs_b[j * 32 + n];
#pragma unroll
      for (int r = 0; r < 8; ++r) {
        float av = c[r] + bias;
        float sig = 1.0f / (1.0f + __expf(-av));
        sfw[r] += sA[w][r + 8 * half][j * 32 + n] * sig;
      }
    }
#pragma unroll
    for (int r = 0; r < 8; ++r) sB[w][r + 8 * half][32 + n] = sfw[r];
  }
  __syncthreads();

  // 6. A-frag of sfw
  v16bf a_sfw;
#pragma unroll
  for (int e = 0; e < 16; ++e) a_sfw[e] = (bf16_t)sB[w][m16][32 + a_col(lane, e)];

  // 7. sf = sfw @ out_fc_W -> sA[:,64:128]; red row -> sA[:,0:64]
#pragma unroll
  for (int nt = 0; nt < 4; ++nt) {
    v16bf b = load_b_frag(out_fc_W, 64, 0, nt * 16, lane);
    v8f c = {};
    c = wmma_bf16(a_sfw, b, c);
    int n = nt * 16 + m16;
#pragma unroll
    for (int r = 0; r < 8; ++r) sA[w][r + 8 * half][64 + n] = c[r];
  }
  {
    const float4* rsrc = (const float4*)(red + (long long)prow * 64 + half * 32);
    float4* rdst = (float4*)(&sA[w][m16][half * 32]);
#pragma unroll
    for (int q = 0; q < 8; ++q) rdst[q] = rsrc[q];
  }
  __syncthreads();

  // 8. h = relu(out_f @ lo1_W) -> sB[:,0:64]   (K = 128)
  v16bf a_of[4];
#pragma unroll
  for (int kf = 0; kf < 4; ++kf)
#pragma unroll
    for (int e = 0; e < 16; ++e)
      a_of[kf][e] = (bf16_t)sA[w][m16][kf * 32 + a_col(lane, e)];
#pragma unroll
  for (int nt = 0; nt < 4; ++nt) {
    v8f c = {};
#pragma unroll
    for (int kf = 0; kf < 4; ++kf) {
      v16bf b = load_b_frag(lo1_W, 64, kf * 32, nt * 16, lane);
      c = wmma_bf16(a_of[kf], b, c);
    }
    int n = nt * 16 + m16;
#pragma unroll
    for (int r = 0; r < 8; ++r) {
      float v = c[r];
      sB[w][r + 8 * half][n] = v > 0.f ? v : 0.f;
    }
  }
  __syncthreads();

  // 9. out = h @ lo2_W + lo2_b
  v16bf a_h[2];
#pragma unroll
  for (int kf = 0; kf < 2; ++kf)
#pragma unroll
    for (int e = 0; e < 16; ++e)
      a_h[kf][e] = (bf16_t)sB[w][m16][kf * 32 + a_col(lane, e)];
  const bool fulltile = (rowbase + 16 <= N_PTS);
#pragma unroll
  for (int nt = 0; nt < 4; ++nt) {
    v8f c = {};
#pragma unroll
    for (int kf = 0; kf < 2; ++kf) {
      v16bf b = load_b_frag(lo2_W, 64, kf * 32, nt * 16, lane);
      c = wmma_bf16(a_h[kf], b, c);
    }
    int n = nt * 16 + m16;
    float bias = lo2_b[n];
    if (fulltile) {
#pragma unroll
      for (int r = 0; r < 8; ++r) {
        int m = rowbase + r + 8 * half;
        out[(long long)m * 64 + n] = c[r] + bias;
      }
    } else {
#pragma unroll
      for (int r = 0; r < 8; ++r) {
        int m = rowbase + r + 8 * half;
        if (m < N_PTS) out[(long long)m * 64 + n] = c[r] + bias;
      }
    }
  }
}

// ---------------------------------------------------------------------------
extern "C" void kernel_launch(void* const* d_in, const int* in_sizes, int n_in,
                              void* d_out, int out_size, void* d_ws, size_t ws_size,
                              hipStream_t stream) {
  const float* feats     = (const float*)d_in[0];
  const int*   coords    = (const int*)d_in[1];
  const float* W_red     = (const float*)d_in[2];
  const float* b_red     = (const float*)d_in[3];
  const float* fc_list_W = (const float*)d_in[4];
  const float* fc_list_b = (const float*)d_in[5];
  const float* fcs_W     = (const float*)d_in[6];
  const float* fcs_b     = (const float*)d_in[7];
  const float* fc_W      = (const float*)d_in[8];
  const float* out_fc_W  = (const float*)d_in[9];
  const float* lo1_W     = (const float*)d_in[10];
  const float* lo2_W     = (const float*)d_in[11];
  const float* lo2_b     = (const float*)d_in[12];
  (void)in_sizes; (void)n_in; (void)out_size; (void)ws_size;

  float* ws   = (float*)d_ws;
  float* red  = ws + RED_OFF;
  float* sums = ws + SUM_OFF;
  float* cnt  = ws + CNT_OFF;
  float* att  = ws + ATT_OFF;
  float* out  = (float*)d_out;

  // sums and cnt are contiguous: zero both in one pass
  long long nz = (long long)TOTSEG * 65;
  k_zero<<<2048, 256, 0, stream>>>(sums, nz);

  int tilesR = (N_PTS + 15) / 16;       // 31250 row tiles
  int blocksR = (tilesR + 7) / 8;       // 8 waves/block
  k_red_scatter<<<blocksR, 256, 0, stream>>>(feats, coords, W_red, b_red, red,
                                             sums, cnt);

  int tilesS = TOTSEG / 16;             // 19382 segment tiles
  int blocksS = (tilesS + 7) / 8;
  k_seg_fc<<<blocksS, 256, 0, stream>>>(fc_list_W, fc_list_b, sums, cnt, att);

  int blocksF = (tilesR + 3) / 4;       // 4 waves/block
  k_final<<<blocksF, 128, 0, stream>>>(coords, fcs_W, fcs_b, fc_W, out_fc_W,
                                       lo1_W, lo2_W, lo2_b, red, att, out);
}
